// CausalSelfAttention_89197880803718
// MI455X (gfx1250) — compile-verified, matrix operand order checked by reading, not run on hardware
//
#include <hip/hip_runtime.h>
#include <hip/hip_bf16.h>

typedef __bf16 bf16;
typedef __attribute__((ext_vector_type(16))) __bf16 v16bf;
typedef __attribute__((ext_vector_type(8)))  __bf16 v8bf;
typedef __attribute__((ext_vector_type(8)))  float  v8f;

#define DEV static __device__ __forceinline__

DEV bf16 f2bf(float f) {
  union { float f; unsigned u; } v; v.f = f;
  unsigned r = v.u + 0x7fffu + ((v.u >> 16) & 1u);   // round-to-nearest-even
  unsigned short h = (unsigned short)(r >> 16);
  bf16 o; __builtin_memcpy(&o, &h, sizeof(o)); return o;
}

DEV v8f zero8() { v8f z;
#pragma unroll
  for (int i = 0; i < 8; i++) z[i] = 0.f;
  return z;
}

DEV v8f wmma_bf16(v16bf a, v16bf b, v8f c) {
  // D(16x16 f32) = A(16x32 bf16) * B(32x16 bf16) + C
  return __builtin_amdgcn_wmma_f32_16x16x32_bf16(false, a, false, b, (short)0, c,
                                                 false, false);
}

// ---- CDNA5 async global->LDS (ASYNCcnt) with safe fallback ----
#if __has_builtin(__builtin_amdgcn_global_load_async_to_lds_b128)
#define ASYNC_LDS 1
typedef __attribute__((ext_vector_type(4))) int v4i;
typedef __attribute__((address_space(1))) v4i as1_v4i;   // global
typedef __attribute__((address_space(3))) v4i as3_v4i;   // LDS
DEV void g2l_b128(const bf16* g, bf16* l) {
  __builtin_amdgcn_global_load_async_to_lds_b128(
      (as1_v4i*)(uintptr_t)g,
      (as3_v4i*)(unsigned)(uintptr_t)l,
      0, 0);
}
DEV void wait_async() {
#if __has_builtin(__builtin_amdgcn_s_wait_asynccnt)
  __builtin_amdgcn_s_wait_asynccnt(0);
#else
  asm volatile("s_wait_asynccnt 0" ::: "memory");
#endif
}
#else
DEV void g2l_b128(const bf16* g, bf16* l) { *(v8bf*)l = *(const v8bf*)g; }
DEV void wait_async() {}
#endif

// Load one 16x32 bf16 WMMA fragment (A layout; also used for B^T rows).
// Lanes 0-15: elems 0-7 = K0-7, elems 8-15 = K16-23
// Lanes 16-31: elems 0-7 = K8-15, elems 8-15 = K24-31
DEV v16bf load_frag(const bf16* base, int stride, int lane) {
  int row  = lane & 15;
  int koff = (lane < 16) ? 0 : 8;
  const bf16* p = base + row * stride + koff;
  v8bf lo = *(const v8bf*)(p);
  v8bf hi = *(const v8bf*)(p + 16);
  return __builtin_shufflevector(lo, hi, 0, 1, 2, 3, 4, 5, 6, 7,
                                 8, 9, 10, 11, 12, 13, 14, 15);
}

// ---------------- fp32 -> bf16 convert ----------------
__global__ void cvt_f32_bf16(const float* __restrict__ src, bf16* __restrict__ dst,
                             int n) {
  int i = (blockIdx.x * blockDim.x + threadIdx.x) * 4;
  if (i + 3 < n) {
    float4 f = *(const float4*)(src + i);
    dst[i + 0] = f2bf(f.x); dst[i + 1] = f2bf(f.y);
    dst[i + 2] = f2bf(f.z); dst[i + 3] = f2bf(f.w);
  }
}

// W is [K][N] row-major fp32; produce Wt = W^T as [N][K] bf16 (B-fragment layout).
__global__ void tcvt_f32_bf16(const float* __restrict__ W, bf16* __restrict__ Wt,
                              int K, int N) {
  int idx = blockIdx.x * blockDim.x + threadIdx.x;   // over N*K, output-contiguous
  if (idx < N * K) {
    int n = idx / K, k = idx % K;
    Wt[idx] = f2bf(W[k * N + n]);
  }
}

// ---------------- 128x128 tiled bf16 GEMM: C = A * Bt^T + bias ----------------
// A: [M][K] bf16 row-major, Bt: [N][K] bf16 row-major.
// MODE 0: scatter into Q/K/V [B,H,T,D] bf16 (fused qkv epilogue)
// MODE 1: write fp32 to fd[M][N]
// Double-buffered LDS pipeline; staging uses async global->LDS when available.
template <int MODE>
__global__ __launch_bounds__(256) void gemm_bf16_128x128(
    const bf16* __restrict__ A, const bf16* __restrict__ Bt,
    const float* __restrict__ bias, int M, int N, int K,
    bf16* __restrict__ qd, bf16* __restrict__ kd, bf16* __restrict__ vd,
    float* __restrict__ fd) {
  __shared__ __align__(16) bf16 As[2][128 * 40];  // 32-K tile, +8 pad per row
  __shared__ __align__(16) bf16 Bs[2][128 * 40];

  const int tid = threadIdx.x;
  const int lane = tid & 31, wave = tid >> 5;
  const int wm = wave >> 1, wn = wave & 1;            // 4x2 wave grid
  const int bm = blockIdx.y * 128, bn = blockIdx.x * 128;

  v8f acc[2][4];
#pragma unroll
  for (int i = 0; i < 2; i++)
#pragma unroll
    for (int j = 0; j < 4; j++) acc[i][j] = zero8();

  auto stage = [&](int buf, int kt) {
#pragma unroll
    for (int i = 0; i < 2; i++) {                     // 512 16B chunks / 256 thr
      int chunk = i * 256 + tid;
      int row = chunk >> 2, c8 = (chunk & 3) * 8;
      g2l_b128(A + (size_t)(bm + row) * K + kt + c8, &As[buf][row * 40 + c8]);
      g2l_b128(Bt + (size_t)(bn + row) * K + kt + c8, &Bs[buf][row * 40 + c8]);
    }
  };

  stage(0, 0);
  wait_async();
  __syncthreads();

  for (int kt = 0; kt < K; kt += 32) {
    const int cur = (kt >> 5) & 1;
    if (kt + 32 < K) stage(cur ^ 1, kt + 32);         // prefetch next tile

    v16bf af[2], bf[4];
#pragma unroll
    for (int mt = 0; mt < 2; mt++)
      af[mt] = load_frag(&As[cur][(wm * 32 + mt * 16) * 40], 40, lane);
#pragma unroll
    for (int nt = 0; nt < 4; nt++)
      bf[nt] = load_frag(&Bs[cur][(wn * 64 + nt * 16) * 40], 40, lane);
#pragma unroll
    for (int mt = 0; mt < 2; mt++)
#pragma unroll
      for (int nt = 0; nt < 4; nt++)
        acc[mt][nt] = wmma_bf16(af[mt], bf[nt], acc[mt][nt]);

    wait_async();
    __syncthreads();
  }

  // Epilogue. C/D layout: elem r -> row (r or r+8), col = lane&15.
  const int colL = lane & 15;
  const int rowOff = (lane < 16) ? 0 : 8;
#pragma unroll
  for (int mt = 0; mt < 2; mt++) {
#pragma unroll
    for (int nt = 0; nt < 4; nt++) {
#pragma unroll
      for (int r = 0; r < 8; r++) {
        int m = bm + wm * 32 + mt * 16 + rowOff + r;
        int n = bn + wn * 64 + nt * 16 + colL;
        float v = acc[mt][nt][r] + bias[n];
        if constexpr (MODE == 0) {
          int b = m >> 11, t = m & 2047;              // T = 2048
          int sect = n >> 10, h = (n & 1023) >> 6, d = n & 63;
          bf16* dst = (sect == 0) ? qd : ((sect == 1) ? kd : vd);
          dst[(((b * 16 + h) * 2048) + t) * 64 + d] = f2bf(v);
        } else {
          fd[(size_t)m * N + n] = v;
        }
      }
    }
  }
}

// ---------------- flash attention ----------------
// Q/K/V: [B,H,T,D] bf16. Output Attn: [B*T, C] bf16 (head-interleaved).
// Block = 4 waves; block owns 64 queries of one (b,h); wave owns 16 queries.
__global__ __launch_bounds__(128) void attn_flash_kernel(
    const bf16* __restrict__ Qb, const bf16* __restrict__ Kb,
    const bf16* __restrict__ Vb, bf16* __restrict__ Attn) {
  const int T = 2048, H = 16;
  __shared__ __align__(16) bf16 Ks[64 * 72];          // [key][d]
  __shared__ __align__(16) bf16 Vs[64 * 72];          // [d][key]  (V^T)
  __shared__ __align__(16) bf16 Ps[4][16 * 72];       // per-wave P tile [q][key]

  int bid = blockIdx.x;
  int qc = bid & 31;                                  // T/64 = 32
  int h  = (bid >> 5) & 15;
  int b  = bid >> 9;
  int qb0 = qc * 64;
  int tid = threadIdx.x, lane = tid & 31, wave = tid >> 5;
  int q0 = qb0 + wave * 16;

  const int bh = b * H + h;
  const bf16* qptr = Qb + (size_t)bh * T * 64;
  const bf16* kptr = Kb + (size_t)bh * T * 64;
  const bf16* vptr = Vb + (size_t)bh * T * 64;

  v16bf qf[2];
#pragma unroll
  for (int ks = 0; ks < 2; ks++) qf[ks] = load_frag(qptr + q0 * 64 + ks * 32, 64, lane);

  float mS[8], lS[8];
  v8f O[4];
#pragma unroll
  for (int r = 0; r < 8; r++) { mS[r] = -3.0e38f; lS[r] = 0.f; }
#pragma unroll
  for (int nd = 0; nd < 4; nd++) O[nd] = zero8();

  const int colL = lane & 15;
  const int rowOff = (lane < 16) ? 0 : 8;

  int nChunks = qb0 / 64 + 1;
  for (int kc = 0; kc < nChunks; kc++) {
    int kb = kc * 64;
    // Stage 64x64 K tile (async) and transposed V tile into LDS.
#pragma unroll
    for (int i = 0; i < 4; i++) {
      int chunk = i * 128 + tid;
      int row = chunk >> 3, c8 = (chunk & 7) * 8;
      g2l_b128(kptr + (kb + row) * 64 + c8, &Ks[row * 72 + c8]);
      v8bf vv = *(const v8bf*)(vptr + (kb + row) * 64 + c8);
#pragma unroll
      for (int j = 0; j < 8; j++) Vs[(c8 + j) * 72 + row] = vv[j];
    }
    wait_async();
    __syncthreads();

    bool diag = (kc == nChunks - 1);
    v8f S[4];
    float rm[8];
#pragma unroll
    for (int r = 0; r < 8; r++) rm[r] = -3.0e38f;

    // S = Q * K^T (4 key sub-tiles, 2 WMMA K-steps each)
#pragma unroll
    for (int n = 0; n < 4; n++) {
      v16bf k0 = load_frag(&Ks[(n * 16) * 72], 72, lane);
      v16bf k1 = load_frag(&Ks[(n * 16) * 72 + 32], 72, lane);
      v8f s = zero8();
      s = wmma_bf16(qf[0], k0, s);
      s = wmma_bf16(qf[1], k1, s);
#pragma unroll
      for (int r = 0; r < 8; r++) {
        float v = s[r] * 0.125f;                      // 1/sqrt(64)
        if (diag) {
          int qi = q0 + rowOff + r;
          int ki = kb + n * 16 + colL;
          if (ki > qi) v = -3.0e38f;                  // causal mask
        }
        s[r] = v;
        rm[r] = fmaxf(rm[r], v);
      }
      S[n] = s;
    }
    // Row max across the 16 lanes of each half-wave.
#pragma unroll
    for (int off = 1; off < 16; off <<= 1)
#pragma unroll
      for (int r = 0; r < 8; r++) rm[r] = fmaxf(rm[r], __shfl_xor(rm[r], off, 32));

    float alpha[8], rs[8];
#pragma unroll
    for (int r = 0; r < 8; r++) {
      float mn = fmaxf(mS[r], rm[r]);
      alpha[r] = __expf(mS[r] - mn);
      mS[r] = mn;
      rs[r] = 0.f;
    }
    // P = exp(S - m); bounce through LDS to re-layout C/D -> A fragment.
#pragma unroll
    for (int n = 0; n < 4; n++) {
#pragma unroll
      for (int r = 0; r < 8; r++) {
        float p = __expf(S[n][r] - mS[r]);
        rs[r] += p;
        Ps[wave][(rowOff + r) * 72 + n * 16 + colL] = f2bf(p);
      }
    }
#pragma unroll
    for (int off = 1; off < 16; off <<= 1)
#pragma unroll
      for (int r = 0; r < 8; r++) rs[r] += __shfl_xor(rs[r], off, 32);
#pragma unroll
    for (int r = 0; r < 8; r++) lS[r] = lS[r] * alpha[r] + rs[r];
#pragma unroll
    for (int nd = 0; nd < 4; nd++)
#pragma unroll
      for (int r = 0; r < 8; r++) O[nd][r] *= alpha[r];

    asm volatile("s_wait_dscnt 0" ::: "memory");      // P stores -> P frag loads

    // O += P * V   (A = P [16 x keys], B^T rows = V^T [d][keys])
#pragma unroll
    for (int nd = 0; nd < 4; nd++) {
#pragma unroll
      for (int ks = 0; ks < 2; ks++) {
        v16bf pf = load_frag(&Ps[wave][ks * 32], 72, lane);
        v16bf vf = load_frag(&Vs[(nd * 16) * 72 + ks * 32], 72, lane);
        O[nd] = wmma_bf16(pf, vf, O[nd]);
      }
    }
    __syncthreads();
  }

  // Normalize and write out as [B*T, C] bf16 with head-interleaved columns.
#pragma unroll
  for (int nd = 0; nd < 4; nd++) {
#pragma unroll
    for (int r = 0; r < 8; r++) {
      float o = O[nd][r] / lS[r];
      int t = q0 + rowOff + r;
      int d = h * 64 + nd * 16 + colL;
      Attn[(size_t)(b * T + t) * 1024 + d] = f2bf(o);
    }
  }
}

// ---------------- host launcher ----------------
extern "C" void kernel_launch(void* const* d_in, const int* in_sizes, int n_in,
                              void* d_out, int out_size, void* d_ws, size_t ws_size,
                              hipStream_t stream) {
  const float* x      = (const float*)d_in[0];
  const float* W_attn = (const float*)d_in[1];
  const float* b_attn = (const float*)d_in[2];
  const float* W_proj = (const float*)d_in[3];
  const float* b_proj = (const float*)d_in[4];
  float* out = (float*)d_out;
  (void)in_sizes; (void)n_in; (void)out_size; (void)ws_size;

  const int B = 4, T = 2048, C = 1024, H = 16;
  const int M = B * T;        // 8192
  const int N1 = 3 * C;       // 3072

  char* ws = (char*)d_ws;
  size_t off = 0;
  auto walloc = [&](size_t bytes) -> void* {
    void* p = ws + off;
    off += (bytes + 255) & ~(size_t)255;
    return p;
  };
  bf16* Xb  = (bf16*)walloc((size_t)M * C * 2);    // x in bf16
  bf16* Wt1 = (bf16*)walloc((size_t)N1 * C * 2);   // W_attn^T bf16
  bf16* Wt2 = (bf16*)walloc((size_t)C * C * 2);    // W_proj^T bf16
  bf16* Qb  = (bf16*)walloc((size_t)M * C * 2);    // [B,H,T,D]
  bf16* Kb  = (bf16*)walloc((size_t)M * C * 2);
  bf16* Vb  = (bf16*)walloc((size_t)M * C * 2);
  bf16* Ab  = (bf16*)walloc((size_t)M * C * 2);    // attention output [M][C]

  cvt_f32_bf16<<<dim3((M * C) / 1024), dim3(256), 0, stream>>>(x, Xb, M * C);
  tcvt_f32_bf16<<<dim3((N1 * C) / 256), dim3(256), 0, stream>>>(W_attn, Wt1, C, N1);
  tcvt_f32_bf16<<<dim3((C * C) / 256), dim3(256), 0, stream>>>(W_proj, Wt2, C, C);

  // QKV GEMM with fused split/transpose-to-heads epilogue
  gemm_bf16_128x128<0><<<dim3(N1 / 128, M / 128), dim3(256), 0, stream>>>(
      Xb, Wt1, b_attn, M, N1, C, Qb, Kb, Vb, nullptr);

  // Flash attention: B*H*(T/64) blocks, 4 waves each
  attn_flash_kernel<<<dim3(B * H * (T / 64)), dim3(128), 0, stream>>>(Qb, Kb, Vb, Ab);

  // Output projection (fp32 result + bias)
  gemm_bf16_128x128<1><<<dim3(C / 128, M / 128), dim3(256), 0, stream>>>(
      Ab, Wt2, b_proj, M, C, C, nullptr, nullptr, nullptr, out);
}
